// CausalSelfAttention_77610059038806
// MI455X (gfx1250) — matrix operand order repaired and checked by reading, not compiled, not run on hardware
//
#include <hip/hip_runtime.h>
#include <hip/hip_bf16.h>

// Causal self-attention forward on gfx1250.
// f16 WMMA (V_WMMA_F32_16X16X32_F16) everywhere, f32 accumulate/softmax.
// Projection GEMMs stage shared B tiles in LDS via GLOBAL_LOAD_ASYNC_TO_LDS
// (ASYNCcnt) with double buffering; B operands batched so waits are partial.

constexpr int BATCH = 2;
constexpr int SEQ   = 2048;
constexpr int EMB   = 1024;
constexpr int NH    = 16;
constexpr int HD    = 64;   // head dim

typedef __attribute__((ext_vector_type(16))) _Float16 v16h;
typedef __attribute__((ext_vector_type(8)))  _Float16 v8h;
typedef __attribute__((ext_vector_type(8)))  float    v8f;

__device__ __forceinline__ v16h cat8(v8h lo, v8h hi) {
    return __builtin_shufflevector(lo, hi, 0,1,2,3,4,5,6,7,8,9,10,11,12,13,14,15);
}

// A-operand (16x32 f16, MxK): lane holds row (lane&15); k chunks at
// kb=(lane>>4)*8 : [kb..kb+7] and [kb+16..kb+23].
__device__ __forceinline__ v16h load_A16(const _Float16* row, int lane) {
    int kb = (lane >> 4) * 8;
    v8h lo = *(const v8h*)(row + kb);
    v8h hi = *(const v8h*)(row + kb + 16);
    return cat8(lo, hi);
}

// B-operand (32x16 f16, KxN): lane holds column (lane&15); k chunks at
// kb=(lane>>4)*8 : [kb..kb+7] and [kb+16..kb+23].
__device__ __forceinline__ v16h load_B16(const _Float16* col, int lane) {
    int kb = (lane >> 4) * 8;
    v8h lo = *(const v8h*)(col + kb);
    v8h hi = *(const v8h*)(col + kb + 16);
    return cat8(lo, hi);
}

__device__ __forceinline__ v8f wmma_f16(v16h a, v16h b, v8f c) {
    return __builtin_amdgcn_wmma_f32_16x16x32_f16(
        false, a, false, b, (short)0, c, false, false);
}

// Async DMA one 16-byte chunk of the shared B tile (32 k x 64 n, stored in LDS
// as 64 rows(n) x 32 halves(k)) from global W^T into LDS.  One chunk/thread.
// GVS addressing: SGPR 64-bit base + 32-bit VGPR byte offset.
__device__ __forceinline__ void issue_B_async(const _Float16* __restrict__ wT,
                                              int n0, int kc,
                                              _Float16* ldsdst, int tid) {
    int n = tid >> 2;        // 0..63 : row of the tile (output column)
    int c = tid & 3;         // 0..3  : 8-half chunk within the 32-k row
    unsigned goff  = (unsigned)((((n0 + n) * EMB) + kc + c * 8) * 2);
    unsigned laddr = (unsigned)(uintptr_t)(ldsdst + n * 32 + c * 8);
    asm volatile("global_load_async_to_lds_b128 %0, %1, %2"
                 :: "v"(laddr), "v"(goff), "s"(wT) : "memory");
}

// ---------------- conversion kernels ----------------

__global__ void __launch_bounds__(256) k_cvt_x(const float* __restrict__ x,
                                               _Float16* __restrict__ xh, int n) {
    int i = blockIdx.x * 256 + threadIdx.x;
    if (i < n) xh[i] = (_Float16)x[i];
}

// wT[n*EMB + k] = (f16) W[k*EMB + n].  blockIdx.y selects which weight.
__global__ void __launch_bounds__(256) k_cvt_wT(const float* __restrict__ wq,
                                                const float* __restrict__ wk,
                                                const float* __restrict__ wv,
                                                const float* __restrict__ wo,
                                                _Float16* __restrict__ wT) {
    const float* w = (blockIdx.y == 0) ? wq : (blockIdx.y == 1) ? wk
                   : (blockIdx.y == 2) ? wv : wo;
    _Float16* o = wT + (size_t)blockIdx.y * EMB * EMB;
    int i = blockIdx.x * 256 + threadIdx.x;       // i = n*EMB + k
    int n = i >> 10;
    int k = i & (EMB - 1);
    o[i] = (_Float16)w[(size_t)k * EMB + n];
}

// ---------------- QKV projection GEMM ----------------
// Block: 8 waves, 128(M) x 64(N); shared B tile double-buffered in LDS via
// async DMA; A rows per-wave, register double-buffered.
__global__ void __launch_bounds__(256) k_qkv(const _Float16* __restrict__ xh,
                                             const _Float16* __restrict__ wTall,
                                             _Float16* __restrict__ Qh,
                                             _Float16* __restrict__ Kh,
                                             _Float16* __restrict__ VT) {
    __shared__ __align__(16) _Float16 bsm[2][64 * 32];
    const int tid  = threadIdx.x;
    const int lane = tid & 31;
    const int wid  = tid >> 5;
    const int l15  = lane & 15;
    const int half = lane >> 4;
    const int mblk = blockIdx.x >> 4;          // 0..31
    const int nblk = blockIdx.x & 15;          // 0..15
    const int which = blockIdx.y;              // 0=Q 1=K 2=V
    const int m0 = mblk * 128 + wid * 16;
    const int n0 = nblk * 64;
    const _Float16* w = wTall + (size_t)which * EMB * EMB;
    const _Float16* arow = xh + (size_t)(m0 + l15) * EMB;

    issue_B_async(w, n0, 0, &bsm[0][0], tid);
    v16h a = load_A16(arow, lane);

    v8f acc0 = {}, acc1 = {}, acc2 = {}, acc3 = {};
    int cur = 0;
    for (int kc = 0; kc < EMB; kc += 32) {
        asm volatile("s_wait_asynccnt 0x0" ::: "memory");
        __syncthreads();                       // buf[cur] visible to all waves
        const int next = kc + 32;
        if (next < EMB) issue_B_async(w, n0, next, &bsm[1 - cur][0], tid);
        v16h anext = a;
        if (next < EMB) anext = load_A16(arow + next, lane);
        const _Float16* bb = &bsm[cur][0];
        // batch all B operand loads so the backend can use partial dscnt waits
        v16h b0 = load_B16(bb + ( 0 + l15) * 32, lane);
        v16h b1 = load_B16(bb + (16 + l15) * 32, lane);
        v16h b2 = load_B16(bb + (32 + l15) * 32, lane);
        v16h b3 = load_B16(bb + (48 + l15) * 32, lane);
        acc0 = wmma_f16(a, b0, acc0);
        acc1 = wmma_f16(a, b1, acc1);
        acc2 = wmma_f16(a, b2, acc2);
        acc3 = wmma_f16(a, b3, acc3);
        a = anext;
        cur ^= 1;
    }

    v8f accs[4] = {acc0, acc1, acc2, acc3};
#pragma unroll
    for (int j = 0; j < 4; ++j) {
#pragma unroll
        for (int r = 0; r < 8; ++r) {
            int m = m0 + r + half * 8;         // global row = b*SEQ + s
            int n = n0 + j * 16 + l15;         // global col = h*HD + d
            int b = m >> 11, s = m & (SEQ - 1);
            int h = n >> 6,  d = n & (HD - 1);
            _Float16 val = (_Float16)accs[j][r];
            if (which == 0)
                Qh[(((size_t)b * NH + h) * SEQ + s) * HD + d] = val;
            else if (which == 1)
                Kh[(((size_t)b * NH + h) * SEQ + s) * HD + d] = val;
            else
                VT[(((size_t)b * NH + h) * HD + d) * SEQ + s] = val;
        }
    }
}

// ---------------- flash attention ----------------
// One wave per 16-row q tile; 8 independent waves/block; grid.y = b*NH+h.
__global__ void __launch_bounds__(256) k_attn(const _Float16* __restrict__ Qh,
                                              const _Float16* __restrict__ Kh,
                                              const _Float16* __restrict__ VT,
                                              _Float16* __restrict__ AO) {
    __shared__ __align__(16) _Float16 pbuf[8][16 * 32];
    const int lane = threadIdx.x & 31;
    const int wid  = threadIdx.x >> 5;
    const int l15  = lane & 15;
    const int half = lane >> 4;
    const int bh = blockIdx.y;
    const int q0 = blockIdx.x * 128 + wid * 16;
    _Float16* pb = &pbuf[wid][0];
    const float scale = 0.125f;                 // 1/sqrt(64)

    const _Float16* qrow = Qh + ((size_t)bh * SEQ + (q0 + l15)) * HD;
    v16h qA0 = load_A16(qrow, lane);            // d = 0..31
    v16h qA1 = load_A16(qrow + 32, lane);       // d = 32..63

    v8f acc[4] = {};
    float m_r[8], l_r[8];
#pragma unroll
    for (int r = 0; r < 8; ++r) { m_r[r] = -3.0e38f; l_r[r] = 0.0f; }

    const int nkb = (q0 + 47) >> 5;             // ceil((q0+16)/32) key blocks
    for (int ib = 0; ib < nkb; ++ib) {
        const int kb = ib * 32;
        // ---- scores: batch all 4 K-tile operands, then 4 WMMAs ----
        const _Float16* kp0 = Kh + ((size_t)bh * SEQ + (kb + l15)) * HD;
        const _Float16* kp1 = kp0 + (size_t)16 * HD;
        v16h kb00 = load_B16(kp0, lane);
        v16h kb01 = load_B16(kp0 + 32, lane);
        v16h kb10 = load_B16(kp1, lane);
        v16h kb11 = load_B16(kp1 + 32, lane);
        v8f c0 = {}, c1 = {};
        c0 = wmma_f16(qA0, kb00, c0);
        c0 = wmma_f16(qA1, kb01, c0);
        c1 = wmma_f16(qA0, kb10, c1);
        c1 = wmma_f16(qA1, kb11, c1);
#pragma unroll
        for (int r = 0; r < 8; ++r) {
            int qi = q0 + r + half * 8;
            int k0 = kb + l15, k1 = k0 + 16;
            float s0 = (float)c0[r] * scale; if (k0 > qi) s0 = -3.0e38f;
            float s1 = (float)c1[r] * scale; if (k1 > qi) s1 = -3.0e38f;
            float mx = fmaxf(s0, s1);
            mx = fmaxf(mx, __shfl_xor(mx, 1, 32));
            mx = fmaxf(mx, __shfl_xor(mx, 2, 32));
            mx = fmaxf(mx, __shfl_xor(mx, 4, 32));
            mx = fmaxf(mx, __shfl_xor(mx, 8, 32));   // within 16-lane half
            float mnew = fmaxf(m_r[r], mx);
            float p0 = __expf(s0 - mnew);
            float p1 = __expf(s1 - mnew);
            float ps = p0 + p1;
            ps += __shfl_xor(ps, 1, 32);
            ps += __shfl_xor(ps, 2, 32);
            ps += __shfl_xor(ps, 4, 32);
            ps += __shfl_xor(ps, 8, 32);
            float corr = __expf(m_r[r] - mnew);
            l_r[r] = l_r[r] * corr + ps;
            m_r[r] = mnew;
#pragma unroll
            for (int j = 0; j < 4; ++j) acc[j][r] *= corr;
            int mrow = r + half * 8;                  // C-layout row
            pb[mrow * 32 + l15]      = (_Float16)p0;
            pb[mrow * 32 + 16 + l15] = (_Float16)p1;
        }
        asm volatile("s_wait_dscnt 0x0" ::: "memory");  // cross-lane LDS relayout
        v16h pA = load_A16(pb + l15 * 32, lane);        // P as A-operand 16x32
        // ---- P @ V: batch all 4 V-tile operands, then 4 WMMAs ----
        const _Float16* vbase = VT + ((size_t)bh * HD + l15) * SEQ + kb;
        v16h vb0 = load_B16(vbase,                    lane);
        v16h vb1 = load_B16(vbase + (size_t)16 * SEQ, lane);
        v16h vb2 = load_B16(vbase + (size_t)32 * SEQ, lane);
        v16h vb3 = load_B16(vbase + (size_t)48 * SEQ, lane);
        acc[0] = wmma_f16(pA, vb0, acc[0]);
        acc[1] = wmma_f16(pA, vb1, acc[1]);
        acc[2] = wmma_f16(pA, vb2, acc[2]);
        acc[3] = wmma_f16(pA, vb3, acc[3]);
    }

    const int b = bh >> 4, h = bh & 15;
#pragma unroll
    for (int j = 0; j < 4; ++j) {
#pragma unroll
        for (int r = 0; r < 8; ++r) {
            int srow = q0 + r + half * 8;
            float v = acc[j][r] / l_r[r];
            AO[((size_t)b * SEQ + srow) * EMB + h * HD + j * 16 + l15] = (_Float16)v;
        }
    }
}

// ---------------- output projection ----------------
__global__ void __launch_bounds__(256) k_oproj(const _Float16* __restrict__ AO,
                                               const _Float16* __restrict__ woT,
                                               float* __restrict__ out) {
    __shared__ __align__(16) _Float16 bsm[2][64 * 32];
    const int tid  = threadIdx.x;
    const int lane = tid & 31;
    const int wid  = tid >> 5;
    const int l15  = lane & 15;
    const int half = lane >> 4;
    const int mblk = blockIdx.x >> 4;
    const int nblk = blockIdx.x & 15;
    const int m0 = mblk * 128 + wid * 16;
    const int n0 = nblk * 64;
    const _Float16* arow = AO + (size_t)(m0 + l15) * EMB;

    issue_B_async(woT, n0, 0, &bsm[0][0], tid);
    v16h a = load_A16(arow, lane);

    v8f acc0 = {}, acc1 = {}, acc2 = {}, acc3 = {};
    int cur = 0;
    for (int kc = 0; kc < EMB; kc += 32) {
        asm volatile("s_wait_asynccnt 0x0" ::: "memory");
        __syncthreads();
        const int next = kc + 32;
        if (next < EMB) issue_B_async(woT, n0, next, &bsm[1 - cur][0], tid);
        v16h anext = a;
        if (next < EMB) anext = load_A16(arow + next, lane);
        const _Float16* bb = &bsm[cur][0];
        v16h b0 = load_B16(bb + ( 0 + l15) * 32, lane);
        v16h b1 = load_B16(bb + (16 + l15) * 32, lane);
        v16h b2 = load_B16(bb + (32 + l15) * 32, lane);
        v16h b3 = load_B16(bb + (48 + l15) * 32, lane);
        acc0 = wmma_f16(a, b0, acc0);
        acc1 = wmma_f16(a, b1, acc1);
        acc2 = wmma_f16(a, b2, acc2);
        acc3 = wmma_f16(a, b3, acc3);
        a = anext;
        cur ^= 1;
    }

    v8f accs[4] = {acc0, acc1, acc2, acc3};
#pragma unroll
    for (int j = 0; j < 4; ++j) {
#pragma unroll
        for (int r = 0; r < 8; ++r) {
            int m = m0 + r + half * 8;
            int n = n0 + j * 16 + l15;
            out[(size_t)m * EMB + n] = accs[j][r];
        }
    }
}

// ---------------- host launch ----------------
extern "C" void kernel_launch(void* const* d_in, const int* in_sizes, int n_in,
                              void* d_out, int out_size, void* d_ws, size_t ws_size,
                              hipStream_t stream) {
    const float* x  = (const float*)d_in[0];
    const float* Wq = (const float*)d_in[1];
    const float* Wk = (const float*)d_in[2];
    const float* Wv = (const float*)d_in[3];
    const float* Wo = (const float*)d_in[4];
    float* out = (float*)d_out;

    const size_t nx = (size_t)BATCH * SEQ * EMB;     // 4,194,304 elems
    _Float16* xh = (_Float16*)d_ws;
    _Float16* wT = xh + nx;                          // 4 transposed weights
    _Float16* Qh = wT + 4ull * EMB * EMB;
    _Float16* Kh = Qh + nx;
    _Float16* VT = Kh + nx;
    _Float16* AO = VT + nx;                          // total ~48 MB

    k_cvt_x<<<dim3((unsigned)(nx / 256)), 256, 0, stream>>>(x, xh, (int)nx);
    k_cvt_wT<<<dim3(EMB * EMB / 256, 4), 256, 0, stream>>>(Wq, Wk, Wv, Wo, wT);
    k_qkv<<<dim3(512, 3), 256, 0, stream>>>(xh, wT, Qh, Kh, VT);
    k_attn<<<dim3(SEQ / 128, BATCH * NH), 256, 0, stream>>>(Qh, Kh, VT, AO);
    k_oproj<<<dim3(512), 256, 0, stream>>>(AO, wT + 3ull * EMB * EMB, out);
}
